// MaskedAttentionBlock_6193342841578
// MI455X (gfx1250) — compile-verified, matrix operand order checked by reading, not run on hardware
//
#include <hip/hip_runtime.h>
#include <hip/hip_bf16.h>

typedef __attribute__((ext_vector_type(16))) _Float16 v16h;
typedef __attribute__((ext_vector_type(8)))  float    v8f;

#define B_DIM 8
#define S_DIM 2048
#define H_DIM 1024

// Per-lane K index for 16-bit WMMA A/B fragments (16x32 A, 32x16 B), wave32:
// lanes 0-15: halfs 0..7 -> K 0..7, halfs 8..15 -> K 16..23 ; lanes 16-31: +8
static __device__ __forceinline__ int kmap(int h, int lane) {
  return (h & 7) + ((h >> 3) << 4) + ((lane >> 4) << 3);
}

// Gather a fragment from LDS laid out [row][k] (row = M for A, N for B), stride ld halfs.
// Contiguous 8-half runs -> compiler emits paired ds_load_b128.
static __device__ __forceinline__ v16h gather_rk(const _Float16* base, int ld, int lane) {
  v16h f;
  const int row = lane & 15;
#pragma unroll
  for (int h = 0; h < 16; ++h) f[h] = base[row * ld + kmap(h, lane)];
  return f;
}

// ---------------------------------------------------------------------------
// Kernel 1: Q/K/V projection  y = x @ W^T  (M=B*S=16384, N=H=1024, K=H=1024)
// fp32 inputs -> f16 LDS tiles -> v_wmma_f32_16x16x32_f16 -> f16 outputs.
// Block = 128 threads (4 waves), tile 128(M) x 64(N), K-step 32.
// Each wave: 32 M-rows x 64 N-cols = 8 accumulators, 8 WMMAs per K-step.
// ---------------------------------------------------------------------------
__global__ __launch_bounds__(128)
void qkv_gemm_f16_kernel(const float* __restrict__ x,
                         const float* __restrict__ wq,
                         const float* __restrict__ wk,
                         const float* __restrict__ wv,
                         _Float16* __restrict__ qws,
                         _Float16* __restrict__ kws,
                         _Float16* __restrict__ vws) {
  __shared__ _Float16 As[128 * 40];  // x tile [128 m][32 k], stride 40 halfs (80B: conflict-free)
  __shared__ _Float16 Bs[64 * 40];   // W tile [64 n][32 k]

  const int n0 = blockIdx.x * 64;
  const int m0 = blockIdx.y * 128;
  const float* W = (blockIdx.z == 0) ? wq : (blockIdx.z == 1) ? wk : wv;
  _Float16* out  = (blockIdx.z == 0) ? qws : (blockIdx.z == 1) ? kws : vws;

  const int tid  = threadIdx.x;
  const int lane = tid & 31;
  const int wid  = tid >> 5;

  v8f acc[8];   // [mt][nt] -> acc[mt*4+nt]
#pragma unroll
  for (int t = 0; t < 8; ++t) acc[t] = (v8f){};

  for (int kc = 0; kc < H_DIM; kc += 32) {
    __syncthreads();
    // Load all tiles into registers first (lets the compiler clause the loads),
    // then convert fp32->f16 and store to LDS.
    float4 va[8], vb[4];
#pragma unroll
    for (int r = 0; r < 8; ++r) {
      const int j = tid + 128 * r;          // 0..1023 float4 slots (128x32 floats)
      va[r] = *(const float4*)&x[(size_t)(m0 + (j >> 3)) * H_DIM + kc + ((j & 7) << 2)];
    }
#pragma unroll
    for (int r = 0; r < 4; ++r) {
      const int j = tid + 128 * r;          // 0..511 float4 slots (64x32 floats)
      vb[r] = *(const float4*)&W[(size_t)(n0 + (j >> 3)) * H_DIM + kc + ((j & 7) << 2)];
    }
#pragma unroll
    for (int r = 0; r < 8; ++r) {
      const int j = tid + 128 * r;
      _Float16* p = &As[(j >> 3) * 40 + ((j & 7) << 2)];
      p[0] = (_Float16)va[r].x; p[1] = (_Float16)va[r].y;
      p[2] = (_Float16)va[r].z; p[3] = (_Float16)va[r].w;
    }
#pragma unroll
    for (int r = 0; r < 4; ++r) {
      const int j = tid + 128 * r;
      _Float16* p = &Bs[(j >> 3) * 40 + ((j & 7) << 2)];
      p[0] = (_Float16)vb[r].x; p[1] = (_Float16)vb[r].y;
      p[2] = (_Float16)vb[r].z; p[3] = (_Float16)vb[r].w;
    }
    __syncthreads();

    v16h a[2];
#pragma unroll
    for (int mt = 0; mt < 2; ++mt)
      a[mt] = gather_rk(&As[(wid * 32 + mt * 16) * 40], 40, lane);
#pragma unroll
    for (int nt = 0; nt < 4; ++nt) {
      const v16h b = gather_rk(&Bs[nt * 16 * 40], 40, lane);  // B[k][n] = W[n][k]
#pragma unroll
      for (int mt = 0; mt < 2; ++mt)
        acc[mt * 4 + nt] = __builtin_amdgcn_wmma_f32_16x16x32_f16(
            false, a[mt], false, b, (short)0, acc[mt * 4 + nt], false, false);
    }
  }

  // C layout: VGPR i -> M = i + 8*(lane>=16), N = lane&15
  const int rb = (lane >> 4) << 3;
#pragma unroll
  for (int mt = 0; mt < 2; ++mt)
#pragma unroll
    for (int nt = 0; nt < 4; ++nt)
#pragma unroll
      for (int i = 0; i < 8; ++i) {
        const int gm = m0 + wid * 32 + mt * 16 + rb + i;
        const int gn = n0 + nt * 16 + (lane & 15);
        out[(size_t)gm * H_DIM + gn] = (_Float16)acc[mt * 4 + nt][i];
      }
}

// ---------------------------------------------------------------------------
// Kernel 2: flash attention (online softmax), causal + attention_mask.
// Block = 256 threads (8 waves) handles one 16-query tile of one batch.
// KV streamed in 32-key tiles, DOUBLE-BUFFERED via global_load_async_to_lds_b128
// (ASYNCcnt path): prefetch tile kt+1 while computing tile kt.
// ---------------------------------------------------------------------------
#define KLD 1032                       // padded K/V LDS row stride (halfs); 2064B, 16B-aligned
#define TILE_BYTES (32 * KLD * 2)      // 66048 B per K or V tile
#define SMEM_SB   (4 * TILE_BYTES)     // 264192 : scores [16 q][32 k] f32
#define SMEM_PB   (SMEM_SB + 16 * 32 * 4)        // 266240 : probs [16 q][32 k] f16
#define SMEM_MB   (SMEM_PB + 16 * 32 * 2)        // 267264 : running max
#define SMEM_LB   (SMEM_MB + 64)                 // running sum
#define SMEM_CB   (SMEM_LB + 64)                 // correction
#define SMEM_TOT  (SMEM_CB + 64)                 // 267456 bytes (< 320KB WGP limit)

// Issue async loads of one 32-key K tile + V tile into LDS double-buffer slot.
// 32 wave-instructions per wave per tile (16 K + 16 V). No static __shared__ in
// this kernel => dynamic LDS starts at byte offset 0, so raw offsets are valid.
static __device__ __forceinline__ void async_load_tile(const _Float16* kbase,
                                                       const _Float16* vbase,
                                                       int kv0, int tid,
                                                       unsigned bufOff) {
#pragma unroll
  for (int r = 0; r < 16; ++r) {
    const int j   = tid + 256 * r;   // 0..4095 uint4 slots (32 rows x 128)
    const int row = j >> 7;
    const int cu  = j & 127;
    const unsigned lk = bufOff + (unsigned)(row * (KLD * 2) + cu * 16);
    const unsigned lv = lk + (unsigned)TILE_BYTES;
    const unsigned go = (unsigned)((kv0 + row) * (H_DIM * 2) + cu * 16);
    asm volatile("global_load_async_to_lds_b128 %0, %1, %2 offset:0"
                 :: "v"(lk), "v"(go), "s"(kbase));
    asm volatile("global_load_async_to_lds_b128 %0, %1, %2 offset:0"
                 :: "v"(lv), "v"(go), "s"(vbase));
  }
}

__global__ __launch_bounds__(256)
void flash_attn_kernel(const _Float16* __restrict__ Qf,
                       const _Float16* __restrict__ Kf,
                       const _Float16* __restrict__ Vf,
                       const int* __restrict__ am,
                       float* __restrict__ out) {
  extern __shared__ char smem[];
  float*    Sb = (float*)   (smem + SMEM_SB);
  _Float16* Pb = (_Float16*)(smem + SMEM_PB);
  float*    mb = (float*)   (smem + SMEM_MB);
  float*    lb = (float*)   (smem + SMEM_LB);
  float*    cb = (float*)   (smem + SMEM_CB);

  const int q0   = blockIdx.x * 16;
  const int b    = blockIdx.y;
  const int tid  = threadIdx.x;
  const int lane = tid & 31;
  const int wid  = tid >> 5;
  const size_t bS = (size_t)b * S_DIM;
  const float NEG_INF = -__builtin_inff();

  const _Float16* kb = Kf + bS * H_DIM;
  const _Float16* vb = Vf + bS * H_DIM;

  // This wave's Q fragments: 16 rows x 128 dims = 4 A-fragments, resident in VGPRs.
  v16h aq[4];
  {
    const int row = lane & 15;
    const _Float16* qrow = &Qf[(bS + q0 + row) * H_DIM + wid * 128];
#pragma unroll
    for (int j = 0; j < 4; ++j)
#pragma unroll
      for (int h = 0; h < 16; ++h)
        aq[j][h] = qrow[j * 32 + kmap(h, lane)];
  }

  v8f acc[8];
#pragma unroll
  for (int t = 0; t < 8; ++t) acc[t] = (v8f){};

  if (tid < 16) { mb[tid] = NEG_INF; lb[tid] = 0.f; cb[tid] = 1.f; }

  const int nTiles = (q0 + 16 + 31) / 32;   // causal early exit

  // Prefetch tile 0 into buffer 0.
  async_load_tile(kb, vb, 0, tid, 0u);

  for (int kt = 0; kt < nTiles; ++kt) {
    const int kv0 = kt * 32;
    const unsigned curOff = (unsigned)(kt & 1) * (2u * TILE_BYTES);
    const unsigned nxtOff = (unsigned)((kt + 1) & 1) * (2u * TILE_BYTES);
    _Float16* Ks = (_Float16*)(smem + curOff);
    _Float16* Vs = (_Float16*)(smem + curOff + TILE_BYTES);

    __syncthreads();   // all waves done reading buf[nxt] (iteration kt-1) & Pb/cb

    if (kt + 1 < nTiles) {
      async_load_tile(kb, vb, kv0 + 32, tid, nxtOff);
      Sb[tid] = 0.f; Sb[tid + 256] = 0.f;
      // 32 newest async ops are the prefetch; <=32 outstanding => tile kt landed.
      asm volatile("s_wait_asynccnt 0x20" ::: "memory");
    } else {
      Sb[tid] = 0.f; Sb[tid + 256] = 0.f;
      asm volatile("s_wait_asynccnt 0x0" ::: "memory");
    }
    __syncthreads();   // every wave's portion of tile kt visible + Sb zeroed

    // Phase A: partial scores S += Q * K^T over this wave's 128-dim chunk.
#pragma unroll
    for (int t = 0; t < 2; ++t) {
      v8f c = (v8f){};
#pragma unroll
      for (int j = 0; j < 4; ++j) {
        v16h bk;   // B[k=h][n=key] gathered from Ks[key][h]; KLD pad -> 16 distinct banks
        const _Float16* krow = &Ks[(t * 16 + (lane & 15)) * KLD + wid * 128 + j * 32];
#pragma unroll
        for (int h = 0; h < 16; ++h) bk[h] = krow[kmap(h, lane)];
        c = __builtin_amdgcn_wmma_f32_16x16x32_f16(
            false, aq[j], false, bk, (short)0, c, false, false);
      }
      const int rb = (lane >> 4) << 3;
#pragma unroll
      for (int i = 0; i < 8; ++i)
        atomicAdd(&Sb[(rb + i) * 32 + t * 16 + (lane & 15)], c[i]);  // ds_add_f32
    }
    __syncthreads();

    // Phase B: masked online softmax; row r owned by thread r.
    if (tid < 16) {
      const int r  = tid;
      const int gq = q0 + r;
      const bool qok = am[bS + gq] != 0;
      const float mold = mb[r];
      float mnew = mold;
      float sc[32];
#pragma unroll
      for (int k = 0; k < 32; ++k) {
        const int gk = kv0 + k;
        const float s = Sb[r * 32 + k] * 0.03125f;   // 1/sqrt(1024)
        const bool ok = qok && (gk <= gq) && (am[bS + gk] != 0);
        sc[k] = ok ? s : NEG_INF;
        if (ok && s > mnew) mnew = s;
      }
      const float corr = (mold == NEG_INF || mnew == NEG_INF) ? 1.f : __expf(mold - mnew);
      float sum = 0.f;
#pragma unroll
      for (int k = 0; k < 32; ++k) {
        const float p = (sc[k] == NEG_INF) ? 0.f : __expf(sc[k] - mnew);
        Pb[r * 32 + k] = (_Float16)p;
        sum += p;
      }
      mb[r] = mnew;
      lb[r] = lb[r] * corr + sum;
      cb[r] = corr;
    }
    __syncthreads();

    // Phase C: rescale accumulators, then O += P * V on this wave's H chunk.
    v16h ap;
    {
      const int row = lane & 15;
#pragma unroll
      for (int h = 0; h < 16; ++h) ap[h] = Pb[row * 32 + kmap(h, lane)];
    }
    const int rb = (lane >> 4) << 3;
#pragma unroll
    for (int nt = 0; nt < 8; ++nt) {
#pragma unroll
      for (int i = 0; i < 8; ++i) acc[nt][i] *= cb[rb + i];
      v16h bv;
      const int hcol = wid * 128 + nt * 16 + (lane & 15);
#pragma unroll
      for (int h = 0; h < 16; ++h) bv[h] = Vs[kmap(h, lane) * KLD + hcol];
      acc[nt] = __builtin_amdgcn_wmma_f32_16x16x32_f16(
          false, ap, false, bv, (short)0, acc[nt], false, false);
    }
  }

  // Epilogue: divide by running sum; fully-masked rows -> 0.
  const int rb = (lane >> 4) << 3;
#pragma unroll
  for (int nt = 0; nt < 8; ++nt) {
    const int hcol = wid * 128 + nt * 16 + (lane & 15);
#pragma unroll
    for (int i = 0; i < 8; ++i) {
      const float ll = lb[rb + i];
      const float v  = (ll > 0.f) ? acc[nt][i] / ll : 0.f;
      out[(bS + q0 + rb + i) * H_DIM + hcol] = v;
    }
  }
}

// ---------------------------------------------------------------------------
extern "C" void kernel_launch(void* const* d_in, const int* in_sizes, int n_in,
                              void* d_out, int out_size, void* d_ws, size_t ws_size,
                              hipStream_t stream) {
  (void)in_sizes; (void)n_in; (void)out_size; (void)ws_size;
  const float* x  = (const float*)d_in[0];
  const int*   am = (const int*)  d_in[1];
  const float* wq = (const float*)d_in[2];
  const float* wk = (const float*)d_in[3];
  const float* wv = (const float*)d_in[4];
  float* out = (float*)d_out;

  const size_t elems = (size_t)B_DIM * S_DIM * H_DIM;   // 16M elements
  _Float16* qws = (_Float16*)d_ws;                       // 32 MB
  _Float16* kws = qws + elems;                           // 32 MB
  _Float16* vws = kws + elems;                           // 32 MB

  dim3 g1(H_DIM / 64, (B_DIM * S_DIM) / 128, 3);
  qkv_gemm_f16_kernel<<<g1, 128, 0, stream>>>(x, wq, wk, wv, qws, kws, vws);

  dim3 g2(S_DIM / 16, B_DIM);
  flash_attn_kernel<<<g2, 256, SMEM_TOT, stream>>>(qws, kws, vws, am, out);
}
